// ComObsAttender_27212912788345
// MI455X (gfx1250) — matrix operand order repaired and checked by reading, not compiled, not run on hardware
//
#include <hip/hip_runtime.h>
#include <cstdint>
#include <cstddef>

// ---------------- problem constants (from reference) ----------------
#define BATCH    1024
#define NAG      27
#define OBS      640
#define GM       (BATCH * NAG)     // 27648 rows
#define GK       OBS               // 640
#define GN       (3 * OBS)         // 1920 cols (q|k|v)

// ---------------- GEMM tiling ----------------
#define TM 128
#define TN 128
#define TK 32
#define LDS_STRIDE 40              // halfs per row (80 bytes, 16B aligned, pads banks)

typedef __attribute__((ext_vector_type(16))) __bf16 v16bf;
typedef __attribute__((ext_vector_type(8)))  __bf16 v8bf;
typedef __attribute__((ext_vector_type(8)))  float  v8f;
typedef __attribute__((ext_vector_type(4)))  int    v4i;

typedef __attribute__((address_space(1))) v4i gas_v4i;   // global (AS1)
typedef __attribute__((address_space(3))) v4i las_v4i;   // LDS (AS3)

// ---------------- CDNA5 async global->LDS copy (ASYNCcnt-tracked) ----------------
#if __has_builtin(__builtin_amdgcn_global_load_async_to_lds_b128)
#define ASYNC_LDS_OK 1
#endif

__device__ __forceinline__ void async_copy16(void* lds_dst, const void* gsrc) {
#ifdef ASYNC_LDS_OK
    __builtin_amdgcn_global_load_async_to_lds_b128(
        (gas_v4i*)(uintptr_t)gsrc,
        (las_v4i*)(unsigned int)(uintptr_t)lds_dst,
        0, 0);
#else
    *(uint4*)lds_dst = *(const uint4*)gsrc;   // fallback: synchronous copy via VGPRs
#endif
}

__device__ __forceinline__ void wait_async0() {
#ifdef ASYNC_LDS_OK
  #if __has_builtin(__builtin_amdgcn_s_wait_asynccnt)
    __builtin_amdgcn_s_wait_asynccnt(0);
  #else
    asm volatile("s_wait_asynccnt 0x0" ::: "memory");
  #endif
#endif
}

// ---------------- fp32 -> bf16 (round to nearest even) ----------------
__global__ void f32_to_bf16(const float* __restrict__ in,
                            unsigned short* __restrict__ out, int n) {
    int i = blockIdx.x * blockDim.x + threadIdx.x;
    if (i < n) {
        unsigned int u = __float_as_uint(in[i]);
        unsigned int r = u + 0x7FFFu + ((u >> 16) & 1u);
        out[i] = (unsigned short)(r >> 16);
    }
}

// ---------------- QKV GEMM:  C[m,n] = sum_k A[m,k]*W[n,k] + bias[n] ----------------
// A (obs) bf16 row-major MxK; Bw (W) bf16 row-major NxK (WMMA-B[k,n] = W[n,k]).
// 256 threads = 8 wave32; each wave computes a 32x64 C tile (2x4 WMMA frags).
__global__ __launch_bounds__(256)
void qkv_gemm_bf16_wmma(const unsigned short* __restrict__ Abf,
                        const unsigned short* __restrict__ Bbf,
                        const float* __restrict__ bias,
                        float* __restrict__ C)
{
    __shared__ unsigned short lA[2][TM * LDS_STRIDE];  // 2 * 10240 B
    __shared__ unsigned short lB[2][TN * LDS_STRIDE];  // 2 * 10240 B

    const int tid = threadIdx.x;
    const int nTilesN = GN / TN;                // 15
    const int m0 = (blockIdx.x / nTilesN) * TM;
    const int n0 = (blockIdx.x % nTilesN) * TN;

    // ---- async tile fill: 4 x b128 per thread per tile (2 A rows + 2 B rows) ----
    const int ldr = tid >> 2;       // 0..63
    const int ldk = (tid & 3) * 8;  // half offset 0,8,16,24

    auto issue_tile = [&](int kb, int buf) {
        const unsigned short* ga0 = Abf + (size_t)(m0 + ldr) * GK + kb + ldk;
        const unsigned short* gb0 = Bbf + (size_t)(n0 + ldr) * GK + kb + ldk;
        async_copy16(&lA[buf][ldr * LDS_STRIDE + ldk],        ga0);
        async_copy16(&lA[buf][(ldr + 64) * LDS_STRIDE + ldk], ga0 + (size_t)64 * GK);
        async_copy16(&lB[buf][ldr * LDS_STRIDE + ldk],        gb0);
        async_copy16(&lB[buf][(ldr + 64) * LDS_STRIDE + ldk], gb0 + (size_t)64 * GK);
        if (kb + 2 * TK <= GK) {                 // speculative L2 prefetch of next tile
            __builtin_prefetch(ga0 + TK, 0, 0);
            __builtin_prefetch(gb0 + TK, 0, 0);
        }
    };

    // ---- per-wave compute mapping: wave tile 32(M) x 64(N) ----
    const int lane = tid & 31;
    const int wave = tid >> 5;          // 0..7
    const int wm = (wave & 3) * 32;     // wave M origin in WG tile
    const int wn = (wave >> 2) * 64;    // wave N origin in WG tile
    const int lr = lane & 15;
    const int lh = lane >> 4;           // lane half

    v8f acc[2][4] = {};

    issue_tile(0, 0);

    int buf = 0;
    for (int kb = 0; kb < GK; kb += TK) {
        wait_async0();        // own wave's fill of lds[buf] complete
        __syncthreads();      // every wave's fill of lds[buf] visible
        if (kb + TK < GK) issue_tile(kb + TK, buf ^ 1);   // overlap next fill w/ WMMA

        // A fragments: 16-bit A 16x32 layout — lane<16 holds K{0..7,16..23},
        // lane>=16 holds K{8..15,24..31}, row M = lane&15.
        v16bf afrag[2], bfrag[4];
        #pragma unroll
        for (int i = 0; i < 2; ++i) {
            const int row = wm + i * 16 + lr;
            union { v16bf v; v8bf h[2]; } u;
            u.h[0] = *(const v8bf*)&lA[buf][row * LDS_STRIDE + lh * 8];
            u.h[1] = *(const v8bf*)&lA[buf][row * LDS_STRIDE + 16 + lh * 8];
            afrag[i] = u.v;
        }
        // B fragments: 16-bit B 32x16 layout — lane&15 = N, lane<16 holds K{0..7,16..23}, lane>=16 holds K{8..15,24..31}.
        #pragma unroll
        for (int j = 0; j < 4; ++j) {
            const int col = wn + j * 16 + lr;
            union { v16bf v; v8bf h[2]; } u;
            u.h[0] = *(const v8bf*)&lB[buf][col * LDS_STRIDE + lh * 8];
            u.h[1] = *(const v8bf*)&lB[buf][col * LDS_STRIDE + 16 + lh * 8];
            bfrag[j] = u.v;
        }

        #pragma unroll
        for (int i = 0; i < 2; ++i)
            #pragma unroll
            for (int j = 0; j < 4; ++j)
                acc[i][j] = __builtin_amdgcn_wmma_f32_16x16x32_bf16(
                    false, afrag[i], false, bfrag[j],
                    (short)0, acc[i][j], false, false);

        __syncthreads();      // lds[buf] free for the fill two iterations out
        buf ^= 1;
    }

    // ---- epilogue: C layout — N = lane&15, M = (lane>>4)*8 + vgpr ----
    #pragma unroll
    for (int i = 0; i < 2; ++i) {
        #pragma unroll
        for (int j = 0; j < 4; ++j) {
            const int mbase = m0 + wm + i * 16 + lh * 8;
            const int col   = n0 + wn + j * 16 + lr;
            const float bv = bias[col];
            #pragma unroll
            for (int r = 0; r < 8; ++r)
                C[(size_t)(mbase + r) * GN + col] = acc[i][j][r] + bv;
        }
    }
}

// ---------------- fused masked attention + concat ----------------
// One 256-thread block (8 wave32) per (batch, agent) row.
__global__ __launch_bounds__(256)
void attn_kernel(const float* __restrict__ obs, const float* __restrict__ qkv,
                 const int* __restrict__ al_idx, const int* __restrict__ al_vis_idx,
                 float* __restrict__ out)
{
    __shared__ float qs[OBS];
    __shared__ float att[32];
    __shared__ int   nb[NAG - 1];
    __shared__ int   visoff[NAG - 1];

    const int row  = blockIdx.x;        // 0..GM-1
    const int b    = row / NAG;
    const int n    = row % NAG;
    const int tid  = threadIdx.x;
    const int lane = tid & 31;
    const int wave = tid >> 5;

    const float* obsrow = obs + (size_t)row * OBS;
    const float* qrow   = qkv + (size_t)row * GN;          // q = cols [0,640)

    for (int d = tid; d < OBS; d += 256) qs[d] = qrow[d];
    if (tid < NAG - 1) {
        nb[tid]     = al_idx[n * (NAG - 1) + tid];
        visoff[tid] = al_vis_idx[tid];
    }
    __syncthreads();

    // stage 1: raw scores q . k_j  (wave-per-neighbor, wave32 shuffle reduce)
    for (int j = wave; j < NAG - 1; j += 8) {
        const float* krow = qkv + ((size_t)b * NAG + nb[j]) * GN + OBS;
        float acc = 0.f;
        for (int d = lane; d < OBS; d += 32) acc += qs[d] * krow[d];
        #pragma unroll
        for (int off = 16; off > 0; off >>= 1) acc += __shfl_xor(acc, off, 32);
        if (lane == 0) att[j] = acc;
    }
    __syncthreads();

    // stage 2: masked softmax * vis on wave 0 (one neighbor per lane)
    if (wave == 0) {
        const bool valid = lane < NAG - 1;
        float vis = 0.f, s = -9999.0f;
        if (valid) {
            vis = (obsrow[visoff[lane]] == 1.0f) ? 1.0f : 0.0f;
            s   = (vis != 0.f) ? att[lane] : -9999.0f;
        }
        float m = s;
        #pragma unroll
        for (int off = 16; off > 0; off >>= 1) m = fmaxf(m, __shfl_xor(m, off, 32));
        const float e = valid ? __expf(s - m) : 0.f;
        float sum = e;
        #pragma unroll
        for (int off = 16; off > 0; off >>= 1) sum += __shfl_xor(sum, off, 32);
        if (valid) att[lane] = (e / sum) * vis;
    }
    __syncthreads();

    // stage 3: env_obs[d] = sum_j att[j] * v_j[d]; write [obs | env_obs]
    float* orow = out + (size_t)row * (2 * OBS);
    for (int d = tid; d < OBS; d += 256) {
        float acc = 0.f;
        #pragma unroll 13
        for (int j = 0; j < NAG - 1; ++j)
            acc += att[j] * qkv[((size_t)b * NAG + nb[j]) * GN + 2 * OBS + d];
        orow[d]       = obsrow[d];
        orow[OBS + d] = acc;
    }
}

// ---------------- host-side launch ----------------
extern "C" void kernel_launch(void* const* d_in, const int* in_sizes, int n_in,
                              void* d_out, int out_size, void* d_ws, size_t ws_size,
                              hipStream_t stream) {
    (void)in_sizes; (void)n_in; (void)out_size; (void)ws_size;

    const float* obs  = (const float*)d_in[0];  // (1024,27,640)
    const float* W    = (const float*)d_in[1];  // (1920,640)
    const float* bias = (const float*)d_in[2];  // (1920,)
    const int* al_idx     = (const int*)d_in[3];  // (27*26,)
    const int* al_vis_idx = (const int*)d_in[4];  // (26,)
    float* out = (float*)d_out;                   // (1024,27,1280)

    // workspace layout: [ qkv f32 | obs bf16 | W bf16 ]
    char* ws = (char*)d_ws;
    float* qkv = (float*)ws;
    size_t off = (size_t)GM * GN * sizeof(float);
    off = (off + 255) & ~(size_t)255;
    unsigned short* obsb = (unsigned short*)(ws + off);
    off += (size_t)GM * GK * sizeof(unsigned short);
    off = (off + 255) & ~(size_t)255;
    unsigned short* Wb = (unsigned short*)(ws + off);

    const int nObs = GM * GK;   // 17,694,720
    const int nW   = GN * GK;   //  1,228,800
    f32_to_bf16<<<(nObs + 255) / 256, 256, 0, stream>>>(obs, obsb, nObs);
    f32_to_bf16<<<(nW + 255) / 256, 256, 0, stream>>>(W, Wb, nW);

    qkv_gemm_bf16_wmma<<<(GM / TM) * (GN / TN), 256, 0, stream>>>(obsb, Wb, bias, qkv);

    attn_kernel<<<GM, 256, 0, stream>>>(obs, qkv, al_idx, al_vis_idx, out);
}
